// Transformer_16295105921364
// MI455X (gfx1250) — compile-verified, matrix operand order checked by reading, not run on hardware
//
#include <hip/hip_runtime.h>
#include <math.h>

// ---------------- Config (T5-Base-like) ----------------
#define C_V   32128
#define C_D   768
#define C_H   12
#define C_HD  64
#define C_M   3072
#define C_L   6
#define C_B   4
#define C_S   512            // SE == SD == 512
#define C_NB  32             // NUM_BUCKETS
#define C_MAXD 128           // MAX_DIST
#define NTOK  (C_B * C_S)    // 2048 tokens per side

// GEMM tiling
#define BM 128               // block tile M (4 waves x 32 rows)
#define BN 64                // block tile N
#define LPAD 40              // LDS row pitch in bf16 (80B: 16B-aligned, bank-spread)

// ---------------- WMMA types ----------------
typedef __attribute__((ext_vector_type(16))) __bf16 v16bf;
typedef __attribute__((ext_vector_type(8)))  __bf16 v8bf;
typedef __attribute__((ext_vector_type(8)))  float  v8f;

__device__ __forceinline__ __bf16 f2bf(float f) {
  // round-to-nearest-even fp32 -> bf16
  unsigned u = __builtin_bit_cast(unsigned, f);
  unsigned r = (u + 0x7FFFu + ((u >> 16) & 1u)) >> 16;
  return __builtin_bit_cast(__bf16, (unsigned short)r);
}

__device__ __forceinline__ void async_g2l_b128(const __bf16* g, void* l) {
  // CDNA5 async global->LDS copy (16B per lane), tracked by ASYNCcnt.
  unsigned laddr = (unsigned)(size_t)l;   // low 32 bits = LDS offset
  asm volatile("global_load_async_to_lds_b128 %0, %1, off"
               :: "v"(laddr), "v"(g) : "memory");
}
__device__ __forceinline__ void wait_async0() {
  asm volatile("s_wait_asynccnt 0x0" ::: "memory");
}

// =====================================================================
// bf16 GEMM: C[M,N] = A[M,K](bf16 row-major) * B, B stored [N,K] bf16.
// Block = 128x64 tile, 4 waves; each wave computes 32x64 (8 WMMAs per
// 32-K step). A/B tiles double-buffered in LDS, filled with async
// global->LDS b128 copies overlapped with WMMA compute.
// blockIdx.z = batch with (z/inner, z%inner) two-level strides.
// Epilogues:
//   0: store fp32          1: ReLU -> store bf16
//   2: fp32 residual add   3: store bf16
//   4: store bf16 transposed into V^T [b,h,hd,s]  (M=NTOK, N=D fixed)
// =====================================================================
template <int EPI>
__global__ __launch_bounds__(128)
void gemm_bf16_wmma(const __bf16* __restrict__ A, long lda,
                    const __bf16* __restrict__ Bm, long ldb,
                    void* __restrict__ Cv, long ldc,
                    const float* __restrict__ Res,
                    int K, int inner,
                    long a_so, long a_si,
                    long b_so, long b_si,
                    long c_so, long c_si) {
  __shared__ __bf16 lA[2][BM][LPAD];
  __shared__ __bf16 lB[2][BN][LPAD];

  const int lane = threadIdx.x;
  const int wave = threadIdx.y;                 // 0..3
  const int tid  = wave * 32 + lane;
  const int half = lane >> 4;                   // 0 or 1
  const int lr   = lane & 15;
  const int tmB = blockIdx.x * BM;
  const int tnB = blockIdx.y * BN;
  const int z  = blockIdx.z;
  const int zo = z / inner, zi = z % inner;
  A  += (long)zo * a_so + (long)zi * a_si;
  Bm += (long)zo * b_so + (long)zi * b_si;
  const long coff = (long)zo * c_so + (long)zi * c_si;

  // ---- async stage of one 32-K tile into LDS buffer `bufi` ----
  auto stage = [&](int bufi, int k0) {
    // A tile: 128 rows x 32 K = 512 x 16B chunks; 4 per thread
#pragma unroll
    for (int t = 0; t < 4; ++t) {
      const int ci = tid + t * 128;
      const int r = ci >> 2, kk = (ci & 3) * 8;
      async_g2l_b128(A + (long)(tmB + r) * lda + k0 + kk, &lA[bufi][r][kk]);
    }
    // B tile: 64 rows x 32 K = 256 x 16B chunks; 2 per thread
#pragma unroll
    for (int t = 0; t < 2; ++t) {
      const int ci = tid + t * 128;
      const int r = ci >> 2, kk = (ci & 3) * 8;
      async_g2l_b128(Bm + (long)(tnB + r) * ldb + k0 + kk, &lB[bufi][r][kk]);
    }
  };

  v8f acc[2][4] = {};
  const int ka = half * 8;        // K run {0..7}/{8..15}
  const int kb = 16 + half * 8;   // K run {16..23}/{24..31}

  stage(0, 0);
  int buf = 0;
  for (int k0 = 0; k0 < K; k0 += 32) {
    wait_async0();
    __syncthreads();
    if (k0 + 32 < K) stage(buf ^ 1, k0 + 32);

    v16bf afr[2];
#pragma unroll
    for (int i = 0; i < 2; ++i) {
      const __bf16* ar = &lA[buf][wave * 32 + i * 16 + lr][0];
      v8bf a0 = *(const v8bf*)(ar + ka);
      v8bf a1 = *(const v8bf*)(ar + kb);
      afr[i] = __builtin_shufflevector(a0, a1, 0, 1, 2, 3, 4, 5, 6, 7,
                                       8, 9, 10, 11, 12, 13, 14, 15);
    }
#pragma unroll
    for (int j = 0; j < 4; ++j) {
      const __bf16* br = &lB[buf][j * 16 + lr][0];
      v8bf b0 = *(const v8bf*)(br + ka);
      v8bf b1 = *(const v8bf*)(br + kb);
      v16bf bfr = __builtin_shufflevector(b0, b1, 0, 1, 2, 3, 4, 5, 6, 7,
                                          8, 9, 10, 11, 12, 13, 14, 15);
      acc[0][j] = __builtin_amdgcn_wmma_f32_16x16x32_bf16(
          false, afr[0], false, bfr, (short)0, acc[0][j], false, false);
      acc[1][j] = __builtin_amdgcn_wmma_f32_16x16x32_bf16(
          false, afr[1], false, bfr, (short)0, acc[1][j], false, false);
    }
    buf ^= 1;
  }

#pragma unroll
  for (int i = 0; i < 2; ++i) {
#pragma unroll
    for (int j = 0; j < 4; ++j) {
      const int col = tnB + j * 16 + lr;
#pragma unroll
      for (int r = 0; r < 8; ++r) {
        const int row = tmB + wave * 32 + i * 16 + half * 8 + r;
        float v = acc[i][j][r];
        if (EPI == 0) {
          ((float*)Cv + coff)[(long)row * ldc + col] = v;
        } else if (EPI == 1) {
          v = v > 0.f ? v : 0.f;
          ((__bf16*)Cv + coff)[(long)row * ldc + col] = f2bf(v);
        } else if (EPI == 2) {
          const long idx = (long)row * ldc + col;
          ((float*)Cv + coff)[idx] = v + (Res + coff)[idx];
        } else if (EPI == 3) {
          ((__bf16*)Cv + coff)[(long)row * ldc + col] = f2bf(v);
        } else {  // EPI == 4: V^T store [b,h,hd,s]; row=b*S+s, col=h*HD+hd
          const int b = row / C_S, s = row % C_S;
          const int h = col / C_HD, kk = col % C_HD;
          ((__bf16*)Cv)[(((long)(b * C_H + h) * C_HD) + kk) * C_S + s] = f2bf(v);
        }
      }
    }
  }
}

// =====================================================================
// Weight transpose + convert: WT[n,k] (bf16) = W[k,n] (fp32)
// =====================================================================
__global__ void wconv_kernel(const float* __restrict__ W,
                             __bf16* __restrict__ WT, int Kd, long total) {
  long i = (long)blockIdx.x * blockDim.x + threadIdx.x;
  if (i >= total) return;
  long n = i / Kd, k = i % Kd;
  long Nd = total / Kd;
  WT[i] = f2bf(W[k * Nd + n]);
}

// =====================================================================
// Embedding gather: out[i,:] = emb[tok[i],:]  (fp32 residual stream)
// =====================================================================
__global__ void embed_kernel(const float* __restrict__ emb,
                             const int* __restrict__ tok,
                             float* __restrict__ out, long total) {
  long i = (long)blockIdx.x * blockDim.x + threadIdx.x;
  if (i >= total) return;
  long row = i / C_D, d = i % C_D;
  out[i] = emb[(long)tok[row] * C_D + d];
}

// =====================================================================
// RMS norm, one wave per token row; writes bf16 (GEMM operand).
// =====================================================================
__global__ void rmsnorm_kernel(const float* __restrict__ x,
                               const float* __restrict__ scale,
                               __bf16* __restrict__ out) {
  const long row = blockIdx.x;
  const int lane = threadIdx.x;
  const float* xr = x + row * C_D;
  float ss = 0.f;
  for (int i = lane; i < C_D; i += 32) { float v = xr[i]; ss += v * v; }
#pragma unroll
  for (int off = 16; off > 0; off >>= 1) ss += __shfl_xor(ss, off, 32);
  const float inv = rsqrtf(ss / (float)C_D + 1e-6f);
  __bf16* orow = out + row * C_D;
  for (int i = lane; i < C_D; i += 32) orow[i] = f2bf(xr[i] * inv * scale[i]);
}

// =====================================================================
// T5 relative-position bias: bias[h,q,s] = rel[h, bucket(s-q)]
// =====================================================================
__global__ void relbias_kernel(const float* __restrict__ rel,
                               float* __restrict__ bias, int bidir) {
  int idx = blockIdx.x * blockDim.x + threadIdx.x;
  if (idx >= C_S * C_S) return;
  const int q = idx / C_S, s = idx % C_S;
  int n = q - s;  // n = -(s - q)
  int bucket = 0, nb = C_NB;
  if (bidir) {
    nb = C_NB / 2;
    if (n < 0) { bucket = nb; n = -n; }
  } else {
    if (n < 0) n = 0;
  }
  const int max_exact = nb / 2;
  int b;
  if (n < max_exact) {
    b = n;
  } else {
    float f = logf((float)n / (float)max_exact) /
              logf((float)C_MAXD / (float)max_exact) * (float)(nb - max_exact);
    b = max_exact + (int)f;
    if (b > nb - 1) b = nb - 1;
  }
  bucket += b;
  for (int h = 0; h < C_H; ++h)
    bias[((long)h * C_S + q) * C_S + s] = rel[h * C_NB + bucket];
}

// =====================================================================
// Fused bias + mask + softmax; reads fp32 scores, writes bf16 probs.
// One wave per (b,h,q) row.
// =====================================================================
__global__ void softmax_kernel(const float* __restrict__ logits,
                               __bf16* __restrict__ probs,
                               const float* __restrict__ bias,
                               const int* __restrict__ qtok,
                               const int* __restrict__ ktok,
                               int causal) {
  const int row = blockIdx.x;                 // (b*H + h)*S + q
  const int q = row % C_S;
  const int h = (row / C_S) % C_H;
  const int b = row / (C_S * C_H);
  const int lane = threadIdx.x;
  const float* L = logits + (long)row * C_S;
  __bf16* P = probs + (long)row * C_S;
  const bool qv = qtok[b * C_S + q] > 0;

  float vals[C_S / 32];
  float maxv = -INFINITY;
#pragma unroll
  for (int i = 0; i < C_S / 32; ++i) {
    const int s = lane + i * 32;
    float v = L[s];
    if (bias) v += bias[((long)h * C_S + q) * C_S + s];
    bool m = qv && (ktok[b * C_S + s] > 0);
    if (causal) m = m && (s <= q);
    v = m ? v : -1e10f;
    vals[i] = v;
    maxv = fmaxf(maxv, v);
  }
#pragma unroll
  for (int off = 16; off > 0; off >>= 1) maxv = fmaxf(maxv, __shfl_xor(maxv, off, 32));
  float sum = 0.f;
#pragma unroll
  for (int i = 0; i < C_S / 32; ++i) { vals[i] = __expf(vals[i] - maxv); sum += vals[i]; }
#pragma unroll
  for (int off = 16; off > 0; off >>= 1) sum += __shfl_xor(sum, off, 32);
  const float inv = 1.f / sum;
#pragma unroll
  for (int i = 0; i < C_S / 32; ++i) P[lane + i * 32] = f2bf(vals[i] * inv);
}

// =====================================================================
// Host-side GEMM dispatch
// =====================================================================
static inline void run_gemm(hipStream_t stream,
                            const __bf16* A, long lda,
                            const __bf16* B, long ldb,
                            void* C, long ldc,
                            const float* Res,
                            int M, int N, int K, int epi,
                            int batch = 1, int inner = 1,
                            long a_so = 0, long a_si = 0,
                            long b_so = 0, long b_si = 0,
                            long c_so = 0, long c_si = 0) {
  dim3 grid(M / BM, N / BN, batch);
  dim3 blk(32, 4);
#define GEMM_CALL(E)                                                           \
  gemm_bf16_wmma<E><<<grid, blk, 0, stream>>>(                                 \
      A, lda, B, ldb, C, ldc, Res, K, inner, a_so, a_si, b_so, b_si, c_so, c_si)
  switch (epi) {
    case 0: GEMM_CALL(0); break;
    case 1: GEMM_CALL(1); break;
    case 2: GEMM_CALL(2); break;
    case 3: GEMM_CALL(3); break;
    default: GEMM_CALL(4); break;
  }
#undef GEMM_CALL
}

// transpose+convert a [K,N] fp32 weight into [N,K] bf16 scratch
static inline void wconv(hipStream_t stream, const float* W, __bf16* WT,
                         int K, int N) {
  long total = (long)K * N;
  wconv_kernel<<<(unsigned)((total + 255) / 256), 256, 0, stream>>>(W, WT, K, total);
}

// =====================================================================
// kernel_launch: full T5 forward
// =====================================================================
extern "C" void kernel_launch(void* const* d_in, const int* in_sizes, int n_in,
                              void* d_out, int out_size, void* d_ws, size_t ws_size,
                              hipStream_t stream) {
  (void)in_sizes; (void)n_in; (void)out_size; (void)ws_size;
  const float* emb      = (const float*)d_in[0];
  const float* rel_enc  = (const float*)d_in[1];
  const float* rel_dec  = (const float*)d_in[2];
  const float* enc_ln1  = (const float*)d_in[3];
  const float* enc_wq   = (const float*)d_in[4];
  const float* enc_wk   = (const float*)d_in[5];
  const float* enc_wv   = (const float*)d_in[6];
  const float* enc_wo   = (const float*)d_in[7];
  const float* enc_ln2  = (const float*)d_in[8];
  const float* enc_wi   = (const float*)d_in[9];
  const float* enc_wmo  = (const float*)d_in[10];
  const float* enc_norm = (const float*)d_in[11];
  const float* dec_ln1  = (const float*)d_in[12];
  const float* dec_sq   = (const float*)d_in[13];
  const float* dec_sk   = (const float*)d_in[14];
  const float* dec_sv   = (const float*)d_in[15];
  const float* dec_so   = (const float*)d_in[16];
  const float* dec_ln2  = (const float*)d_in[17];
  const float* dec_cq   = (const float*)d_in[18];
  const float* dec_ck   = (const float*)d_in[19];
  const float* dec_cv   = (const float*)d_in[20];
  const float* dec_co   = (const float*)d_in[21];
  const float* dec_ln3  = (const float*)d_in[22];
  const float* dec_wi   = (const float*)d_in[23];
  const float* dec_wmo  = (const float*)d_in[24];
  const float* dec_norm = (const float*)d_in[25];
  const float* logits_w = (const float*)d_in[26];
  const int* enc_tokens     = (const int*)d_in[27];
  const int* dec_in_tokens  = (const int*)d_in[28];
  const int* dec_tgt_tokens = (const int*)d_in[29];

  // ---- workspace bump allocator (256B aligned) ----
  size_t off = 0;
  auto allocB = [&](size_t bytes) {
    void* p = (char*)d_ws + off;
    off += (bytes + 255) & ~(size_t)255;
    return p;
  };
  float*  enc_x   = (float*)allocB((size_t)NTOK * C_D * 4);
  float*  dec_x   = (float*)allocB((size_t)NTOK * C_D * 4);
  float*  scores  = (float*)allocB((size_t)C_B * C_H * C_S * C_S * 4);
  float*  bias_e  = (float*)allocB((size_t)C_H * C_S * C_S * 4);
  float*  bias_d  = (float*)allocB((size_t)C_H * C_S * C_S * 4);
  __bf16* h_bf    = (__bf16*)allocB((size_t)NTOK * C_D * 2);
  __bf16* q_bf    = (__bf16*)allocB((size_t)NTOK * C_D * 2);
  __bf16* k_bf    = (__bf16*)allocB((size_t)NTOK * C_D * 2);
  __bf16* vT_bf   = (__bf16*)allocB((size_t)C_B * C_H * C_HD * C_S * 2);
  __bf16* ao_bf   = (__bf16*)allocB((size_t)NTOK * C_D * 2);
  __bf16* encd_bf = (__bf16*)allocB((size_t)NTOK * C_D * 2);
  __bf16* mlph_bf = (__bf16*)allocB((size_t)NTOK * C_M * 2);
  __bf16* prob_bf = (__bf16*)allocB((size_t)C_B * C_H * C_S * C_S * 2);
  __bf16* wT      = (__bf16*)allocB((size_t)C_V * C_D * 2);  // weight scratch

  const long DHH = (long)C_D * C_H * C_HD;  // 768*768 per-layer slab
  const int  BH  = C_B * C_H;               // 48 attention batches
  const long aS  = (long)C_S * C_D;         // per-b stride in [token,768]
  const long sS  = (long)C_S * C_S;

  // ---- relative position biases ----
  {
    int nt = C_S * C_S;
    relbias_kernel<<<(nt + 255) / 256, 256, 0, stream>>>(rel_enc, bias_e, 1);
    relbias_kernel<<<(nt + 255) / 256, 256, 0, stream>>>(rel_dec, bias_d, 0);
  }
  // ---- embeddings ----
  {
    long total = (long)NTOK * C_D;
    unsigned g = (unsigned)((total + 255) / 256);
    embed_kernel<<<g, 256, 0, stream>>>(emb, enc_tokens, enc_x, total);
    embed_kernel<<<g, 256, 0, stream>>>(emb, dec_in_tokens, dec_x, total);
  }

  // QKV projections from bf16 activations `src` (row-major [NTOK, D])
  auto qkv = [&](const __bf16* src, const float* wq_l, const float* wk_l,
                 const float* wv_l) {
    wconv(stream, wq_l, wT, C_D, C_D);
    run_gemm(stream, src, C_D, wT, C_D, q_bf, C_D, nullptr, NTOK, C_D, C_D, 3);
    wconv(stream, wk_l, wT, C_D, C_D);
    run_gemm(stream, src, C_D, wT, C_D, k_bf, C_D, nullptr, NTOK, C_D, C_D, 3);
    wconv(stream, wv_l, wT, C_D, C_D);
    run_gemm(stream, src, C_D, wT, C_D, vT_bf, 0, nullptr, NTOK, C_D, C_D, 4);
  };

  // full attention block given q_bf/k_bf/vT_bf
  auto attention = [&](const float* bias, const int* qtok, const int* ktok,
                       int causal, const float* wo_l, float* xres) {
    // scores = Q . K^T  (k_bf row-major [token,D] is exactly [N,K] per head)
    run_gemm(stream, q_bf, C_D, k_bf, C_D, scores, C_S, nullptr,
             C_S, C_S, C_HD, 0, BH, C_H,
             aS, C_HD, aS, C_HD, (long)C_H * sS, sS);
    softmax_kernel<<<C_B * C_H * C_S, 32, 0, stream>>>(scores, prob_bf, bias,
                                                       qtok, ktok, causal);
    // attnout = P . V  via V^T [b,h,hd,s]  -> store bf16 row-major [token,D]
    run_gemm(stream, prob_bf, C_S, vT_bf, C_S, ao_bf, C_D, nullptr,
             C_S, C_HD, C_S, 3, BH, C_H,
             (long)C_H * sS, sS, (long)C_H * C_HD * C_S, (long)C_HD * C_S,
             aS, C_HD);
    // x += attnout . Wo
    wconv(stream, wo_l, wT, C_D, C_D);
    run_gemm(stream, ao_bf, C_D, wT, C_D, xres, C_D, xres, NTOK, C_D, C_D, 2);
  };

  auto mlp = [&](float* x, const float* ln, const float* wi_l,
                 const float* wmo_l) {
    rmsnorm_kernel<<<NTOK, 32, 0, stream>>>(x, ln, h_bf);
    wconv(stream, wi_l, wT, C_D, C_M);
    run_gemm(stream, h_bf, C_D, wT, C_D, mlph_bf, C_M, nullptr, NTOK, C_M, C_D, 1);
    wconv(stream, wmo_l, wT, C_M, C_D);
    run_gemm(stream, mlph_bf, C_M, wT, C_M, x, C_D, x, NTOK, C_D, C_M, 2);
  };

  // ================= Encoder =================
  for (int l = 0; l < C_L; ++l) {
    rmsnorm_kernel<<<NTOK, 32, 0, stream>>>(enc_x, enc_ln1 + (long)l * C_D, h_bf);
    qkv(h_bf, enc_wq + l * DHH, enc_wk + l * DHH, enc_wv + l * DHH);
    attention(bias_e, enc_tokens, enc_tokens, 0, enc_wo + l * DHH, enc_x);
    mlp(enc_x, enc_ln2 + (long)l * C_D, enc_wi + (long)l * C_D * C_M,
        enc_wmo + (long)l * C_M * C_D);
  }
  rmsnorm_kernel<<<NTOK, 32, 0, stream>>>(enc_x, enc_norm, encd_bf);

  // ================= Decoder =================
  for (int l = 0; l < C_L; ++l) {
    // self attention (causal + dec bias)
    rmsnorm_kernel<<<NTOK, 32, 0, stream>>>(dec_x, dec_ln1 + (long)l * C_D, h_bf);
    qkv(h_bf, dec_sq + l * DHH, dec_sk + l * DHH, dec_sv + l * DHH);
    attention(bias_d, dec_tgt_tokens, dec_tgt_tokens, 1, dec_so + l * DHH, dec_x);
    // cross attention (no bias, K/V from encoder output)
    rmsnorm_kernel<<<NTOK, 32, 0, stream>>>(dec_x, dec_ln2 + (long)l * C_D, h_bf);
    wconv(stream, dec_cq + l * DHH, wT, C_D, C_D);
    run_gemm(stream, h_bf, C_D, wT, C_D, q_bf, C_D, nullptr, NTOK, C_D, C_D, 3);
    wconv(stream, dec_ck + l * DHH, wT, C_D, C_D);
    run_gemm(stream, encd_bf, C_D, wT, C_D, k_bf, C_D, nullptr, NTOK, C_D, C_D, 3);
    wconv(stream, dec_cv + l * DHH, wT, C_D, C_D);
    run_gemm(stream, encd_bf, C_D, wT, C_D, vT_bf, 0, nullptr, NTOK, C_D, C_D, 4);
    attention(nullptr, dec_tgt_tokens, enc_tokens, 0, dec_co + l * DHH, dec_x);
    // MLP
    mlp(dec_x, dec_ln3 + (long)l * C_D, dec_wi + (long)l * C_D * C_M,
        dec_wmo + (long)l * C_M * C_D);
  }
  rmsnorm_kernel<<<NTOK, 32, 0, stream>>>(dec_x, dec_norm, h_bf);

  // ================= Logits =================
  wconv(stream, logits_w, wT, C_D, C_V);
  run_gemm(stream, h_bf, C_D, wT, C_D, (float*)d_out, C_V, nullptr,
           NTOK, C_V, C_D, 0);
}
